// LIIF_74869869904177
// MI455X (gfx1250) — matrix operand order repaired and checked by reading, not compiled, not run on hardware
//
#include <hip/hip_runtime.h>

typedef _Float16 f16;
typedef __attribute__((ext_vector_type(16))) _Float16 v16h;
typedef __attribute__((ext_vector_type(8)))  _Float16 v8h;
typedef __attribute__((ext_vector_type(8)))  float    v8f;
typedef __attribute__((ext_vector_type(4)))  int      v4i;

#define HH    96
#define WW    96
#define CC    64
#define NCIN  3
#define CU    576
#define INDIM 580
#define HID   256
#define QTOT  65536
#define RTOT  (4 * QTOT)

#define MTILE 32      // rows per workgroup (2 WMMA M-subtiles)
#define KT0   19      // ceil(580/32) K-steps for layer 0 (K padded to 608)
#define KTH   8       // 256/32 K-steps for hidden layers
#define S0    624     // LDS stride (halfs) for X0 tile (608 + 16 pad, 16B-aligned rows)
#define SH    272     // LDS stride (halfs) for hidden tiles (256 + 16 pad)
#define CPR   78      // 16B copy-chunks per row: 9*8 feature + 1 coord + 5 zero-pad

#if defined(__has_builtin)
#if __has_builtin(__builtin_amdgcn_global_load_async_to_lds_b128) && \
    __has_builtin(__builtin_amdgcn_s_wait_asynccnt)
#define USE_ASYNC_GATHER 1
#endif
#endif

#define AS1 __attribute__((address_space(1)))
#define AS3 __attribute__((address_space(3)))

// ---------------------------------------------------------------------------
// Encoder: 3x3 SAME conv [3,96,96] -> channel-last f16 [96*96][64]
// ---------------------------------------------------------------------------
__global__ void enc_conv(const float* __restrict__ inp, const float* __restrict__ w,
                         const float* __restrict__ b, f16* __restrict__ feat_cl) {
    int idx = blockIdx.x * 256 + threadIdx.x;      // idx = (y*96+x)*64 + c
    if (idx >= CC * HH * WW) return;
    int c   = idx & 63;
    int pix = idx >> 6;
    int y = pix / WW, x = pix % WW;
    float acc = b[c];
#pragma unroll
    for (int ci = 0; ci < NCIN; ++ci)
#pragma unroll
        for (int ki = 0; ki < 3; ++ki) {
            int yy = y + ki - 1;
            if (yy < 0 || yy >= HH) continue;
#pragma unroll
            for (int kj = 0; kj < 3; ++kj) {
                int xx = x + kj - 1;
                if (xx < 0 || xx >= WW) continue;
                acc += inp[(ci * HH + yy) * WW + xx] * w[((c * NCIN + ci) * 3 + ki) * 3 + kj];
            }
        }
    feat_cl[idx] = (f16)acc;
}

// ---------------------------------------------------------------------------
// Weight pre-swizzle into per-lane WMMA B-fragment layout:
//   dst[((nt*KT + kt)*32 + lane)*16 + i] corresponds to
//   W[perm(kt*32 + (lane>>4)*16 + i)][nt*16 + (lane&15)]
// For layer 0 the K axis is permuted to patch-position-major (k' = kpos*64+c
// maps to original unfold row c*9+kpos) to match the vectorized gather.
// ---------------------------------------------------------------------------
__global__ void wswz(const float* __restrict__ src, f16* __restrict__ dst,
                     int Kreal, int Nreal, int KT, int NT, int permute0) {
    int idx = blockIdx.x * 256 + threadIdx.x;
    int total = KT * NT * 512;
    if (idx >= total) return;
    int nt = idx / (KT * 512);
    int r  = idx % (KT * 512);
    int kt = r / 512;
    int e  = r % 512;
    int lane = e / 16, i = e % 16;
    int n = nt * 16 + (lane & 15);
    int k = kt * 32 + ((lane >> 4) << 4) + i;      // padded (possibly permuted) K index
    int korig;
    if (permute0 && k < CU) {
        int kpos = k >> 6, c = k & 63;             // k' = kpos*64 + c
        korig = c * 9 + kpos;                      // unfold order is c-major, patch-minor
    } else {
        korig = k;
    }
    float v = (korig < Kreal && n < Nreal) ? src[korig * Nreal + n] : 0.0f;
    dst[idx] = (f16)v;
}

// ---------------------------------------------------------------------------
// A-fragment loader from LDS (16-bit A 16x32, this code's contiguous k order):
//   lane m (0..15):  K = hi*16 + 0..15 (contiguous, matches wswz B packing)
// -> two contiguous 16B LDS reads per lane.
// ---------------------------------------------------------------------------
__device__ __forceinline__ v16h load_a(const f16* xs, int stride, int lane, int kt) {
    int m = lane & 15, hi = lane >> 4;
    const f16* row = xs + m * stride + kt * 32;
    v8h lo = *(const v8h*)(row + hi * 16);
    v8h hv = *(const v8h*)(row + hi * 16 + 8);
    v16h a;
#pragma unroll
    for (int i = 0; i < 8; ++i) { a[i] = lo[i]; a[i + 8] = hv[i]; }
    return a;
}

// One hidden layer over the 32-row tile: X[32 x K] @ W[K x 256] + b, ReLU.
// Wave w owns N-tiles {2w, 2w+1} x M-subtiles {0,1}: 4 accumulators; each
// B fragment feeds 2 WMMAs and each A fragment feeds 2 WMMAs.
template <int KT>
__device__ __forceinline__ void layer(const f16* __restrict__ xsrc, int sstride,
                                      f16* __restrict__ xdst, int dstride,
                                      const f16* __restrict__ wsz,
                                      const float* __restrict__ bias,
                                      int wave, int lane) {
    int nt0 = wave * 2, nt1 = wave * 2 + 1;
    int n0 = nt0 * 16 + (lane & 15);
    int n1 = nt1 * 16 + (lane & 15);
    float b0v = bias[n0], b1v = bias[n1];
    v8f acc00, acc01, acc10, acc11;
#pragma unroll
    for (int r = 0; r < 8; ++r) {
        acc00[r] = b0v; acc01[r] = b1v;
        acc10[r] = b0v; acc11[r] = b1v;
    }
#pragma unroll
    for (int kt = 0; kt < KT; ++kt) {
        v16h a0 = load_a(xsrc, sstride, lane, kt);
        v16h a1 = load_a(xsrc + 16 * sstride, sstride, lane, kt);
        v16h f0 = *(const v16h*)(wsz + (((size_t)nt0 * KT + kt) * 32 + lane) * 16);
        v16h f1 = *(const v16h*)(wsz + (((size_t)nt1 * KT + kt) * 32 + lane) * 16);
        acc00 = __builtin_amdgcn_wmma_f32_16x16x32_f16(false, a0, false, f0, (short)0, acc00, false, false);
        acc01 = __builtin_amdgcn_wmma_f32_16x16x32_f16(false, a0, false, f1, (short)0, acc01, false, false);
        acc10 = __builtin_amdgcn_wmma_f32_16x16x32_f16(false, a1, false, f0, (short)0, acc10, false, false);
        acc11 = __builtin_amdgcn_wmma_f32_16x16x32_f16(false, a1, false, f1, (short)0, acc11, false, false);
    }
    // D layout: VGPR r -> M = r (+8 for lanes 16..31), N = lane&15
    int mb = (lane >> 4) << 3;
#pragma unroll
    for (int r = 0; r < 8; ++r) {
        int m = r + mb;
        xdst[m * dstride + n0]        = (f16)fmaxf(acc00[r], 0.0f);
        xdst[m * dstride + n1]        = (f16)fmaxf(acc01[r], 0.0f);
        xdst[(m + 16) * dstride + n0] = (f16)fmaxf(acc10[r], 0.0f);
        xdst[(m + 16) * dstride + n1] = (f16)fmaxf(acc11[r], 0.0f);
    }
}

// ---------------------------------------------------------------------------
// Fused gather + 5-layer MLP. One workgroup = 32 rows (row = j*Q + q).
// LDS: X0 tile (dead after layer0) is aliased by the hidden ping buffer.
// Gather uses the CDNA5 async global->LDS engine when available (ASYNCcnt).
// ---------------------------------------------------------------------------
__global__ void __launch_bounds__(256) mlp_kernel(
    const f16* __restrict__ feat,
    const float* __restrict__ coord, const float* __restrict__ cell,
    const f16* __restrict__ w0s, const f16* __restrict__ w1s,
    const f16* __restrict__ w2s, const f16* __restrict__ w3s,
    const f16* __restrict__ w4s,
    const float* __restrict__ b0, const float* __restrict__ b1,
    const float* __restrict__ b2, const float* __restrict__ b3,
    const float* __restrict__ b4,
    float* __restrict__ preds, float* __restrict__ areas) {
    __shared__ __align__(16) f16 smem[MTILE * S0 + MTILE * SH];  // 57,344 B
    f16* sX0 = smem;                 // 32 x 624
    f16* sXA = smem + MTILE * S0;    // 32 x 272
    f16* sXB = smem;                 // aliases sX0 (dead after layer 0)
    __shared__ int   s_iy[MTILE], s_ix[MTILE];
    __shared__ float s_r0[MTILE], s_r1[MTILE], s_rc0[MTILE], s_rc1[MTILE];

    int tid  = threadIdx.x;
    int lane = tid & 31, wave = tid >> 5;
    int grBase = blockIdx.x * MTILE;

    if (tid == 0) {  // warm the weight path in L2 (global_prefetch_b8)
        __builtin_prefetch(w0s, 0, 2);
        __builtin_prefetch(w1s, 0, 2);
        __builtin_prefetch(w2s, 0, 2);
        __builtin_prefetch(w3s, 0, 2);
        __builtin_prefetch(w4s, 0, 2);
    }

    if (tid < MTILE) {
        int gr = grBase + tid;
        int j  = gr >> 16;            // ensemble tap index (vx outer, vy inner)
        int q  = gr & (QTOT - 1);
        float vx = (j & 2) ? 1.0f : -1.0f;
        float vy = (j & 1) ? 1.0f : -1.0f;
        float c0 = coord[q * 2 + 0], c1 = coord[q * 2 + 1];
        const float rr = 1.0f / 96.0f, bnd = 1.0f - 1e-6f;
        float cc0 = fminf(fmaxf(c0 + vx * rr + 1e-6f, -bnd), bnd);
        float cc1 = fminf(fmaxf(c1 + vy * rr + 1e-6f, -bnd), bnd);
        int iy = (int)fminf(fmaxf(rintf((cc0 + 1.0f) * 48.0f - 0.5f), 0.0f), 95.0f);
        int ix = (int)fminf(fmaxf(rintf((cc1 + 1.0f) * 48.0f - 0.5f), 0.0f), 95.0f);
        float rel0 = (c0 + 1.0f) * 96.0f - (float)(2 * iy + 1);
        float rel1 = (c1 + 1.0f) * 96.0f - (float)(2 * ix + 1);
        s_iy[tid] = iy; s_ix[tid] = ix;
        s_r0[tid] = rel0; s_r1[tid] = rel1;
        s_rc0[tid] = cell[q * 2 + 0] * 96.0f;
        s_rc1[tid] = cell[q * 2 + 1] * 96.0f;
        areas[gr] = fabsf(rel0 * rel1) + 1e-9f;
    }
    __syncthreads();

    // Vectorized gather: X0 K-order is patch-position-major, so each of the 9
    // patch positions is a contiguous 64-channel (128 B) segment of channel-
    // last feat -> 16 B chunks, coalesced 128 B per (row, position).
    for (int idx = tid; idx < MTILE * CPR; idx += 256) {
        int row = idx / CPR, cj = idx % CPR;
        f16* ldst;
        if (cj < 72) {
            int k = cj >> 3, sub = cj & 7;
            ldst = sX0 + row * S0 + (k << 6) + (sub << 3);
            int yy = s_iy[row] + (k / 3) - 1;
            int xx = s_ix[row] + (k % 3) - 1;
            if (yy >= 0 && yy < HH && xx >= 0 && xx < WW) {
                const f16* gsrc = feat + (((size_t)(yy * WW + xx)) << 6) + (sub << 3);
#ifdef USE_ASYNC_GATHER
                __builtin_amdgcn_global_load_async_to_lds_b128(
                    (AS1 v4i*)gsrc, (AS3 v4i*)ldst, 0, 0);
#else
                *(v8h*)ldst = *(const v8h*)gsrc;
#endif
            } else {
                v8h z = {};
                *(v8h*)ldst = z;
            }
        } else {
            ldst = sX0 + row * S0 + 576 + ((cj - 72) << 3);
            v8h v = {};
            if (cj == 72) {
                v[0] = (f16)s_r0[row]; v[1] = (f16)s_r1[row];
                v[2] = (f16)s_rc0[row]; v[3] = (f16)s_rc1[row];
            }
            *(v8h*)ldst = v;
        }
    }
#ifdef USE_ASYNC_GATHER
    __builtin_amdgcn_s_wait_asynccnt(0);   // drain ASYNCcnt before sharing the tile
#endif
    __syncthreads();

    layer<KT0>(sX0, S0, sXA, SH, w0s, b0, wave, lane);
    __syncthreads();
    layer<KTH>(sXA, SH, sXB, SH, w1s, b1, wave, lane);   // sXB overlays dead sX0
    __syncthreads();
    layer<KTH>(sXB, SH, sXA, SH, w2s, b2, wave, lane);
    __syncthreads();
    layer<KTH>(sXA, SH, sXB, SH, w3s, b3, wave, lane);
    __syncthreads();

    if (wave < 2) {  // final 256 -> 3 (padded to 16): wave w takes M-subtile w
        int n = lane & 15;
        float bv = (n < 3) ? b4[n] : 0.0f;
        v8f acc;
#pragma unroll
        for (int r = 0; r < 8; ++r) acc[r] = bv;
        const f16* xsrc = sXB + wave * 16 * SH;
#pragma unroll
        for (int kt = 0; kt < KTH; ++kt) {
            v16h a  = load_a(xsrc, SH, lane, kt);
            v16h bf = *(const v16h*)(w4s + ((size_t)(kt * 32 + lane)) * 16);
            acc = __builtin_amdgcn_wmma_f32_16x16x32_f16(false, a, false, bf, (short)0, acc, false, false);
        }
        if (n < 3) {
            int mb = (lane >> 4) << 3;
#pragma unroll
            for (int r = 0; r < 8; ++r) {
                int gr = grBase + wave * 16 + mb + r;
                preds[gr * 3 + n] = acc[r];
            }
        }
    }
}

// ---------------------------------------------------------------------------
// Local-ensemble blend with diagonal area swap: weight(pred_j) = area_{3-j}/tot
// ---------------------------------------------------------------------------
__global__ void combine(const float* __restrict__ preds, const float* __restrict__ areas,
                        float* __restrict__ out) {
    int q = blockIdx.x * 256 + threadIdx.x;
    if (q >= QTOT) return;
    float a0 = areas[q], a1 = areas[QTOT + q], a2 = areas[2 * QTOT + q], a3 = areas[3 * QTOT + q];
    float inv = 1.0f / (a0 + a1 + a2 + a3);
    float wgt0 = a3 * inv, wgt1 = a2 * inv, wgt2 = a1 * inv, wgt3 = a0 * inv;
#pragma unroll
    for (int c = 0; c < 3; ++c) {
        float v = preds[(0 * QTOT + q) * 3 + c] * wgt0 +
                  preds[(1 * QTOT + q) * 3 + c] * wgt1 +
                  preds[(2 * QTOT + q) * 3 + c] * wgt2 +
                  preds[(3 * QTOT + q) * 3 + c] * wgt3;
        out[q * 3 + c] = v;
    }
}

extern "C" void kernel_launch(void* const* d_in, const int* in_sizes, int n_in,
                              void* d_out, int out_size, void* d_ws, size_t ws_size,
                              hipStream_t stream) {
    const float* inp   = (const float*)d_in[0];
    const float* coord = (const float*)d_in[1];
    const float* cell  = (const float*)d_in[2];
    const float* enc_w = (const float*)d_in[3];
    const float* enc_b = (const float*)d_in[4];
    const float* w0 = (const float*)d_in[5];  const float* b0 = (const float*)d_in[6];
    const float* w1 = (const float*)d_in[7];  const float* b1 = (const float*)d_in[8];
    const float* w2 = (const float*)d_in[9];  const float* b2 = (const float*)d_in[10];
    const float* w3 = (const float*)d_in[11]; const float* b3 = (const float*)d_in[12];
    const float* w4 = (const float*)d_in[13]; const float* b4 = (const float*)d_in[14];
    float* out = (float*)d_out;

    char* ws = (char*)d_ws;
    size_t off = 0;
    f16* feat = (f16*)(ws + off); off += (size_t)CC * HH * WW * 2;            // 1,179,648
    f16* w0s  = (f16*)(ws + off); off += (size_t)KT0 * 16 * 512 * 2;          //   311,296
    f16* w1s  = (f16*)(ws + off); off += (size_t)KTH * 16 * 512 * 2;          //   131,072
    f16* w2s  = (f16*)(ws + off); off += (size_t)KTH * 16 * 512 * 2;
    f16* w3s  = (f16*)(ws + off); off += (size_t)KTH * 16 * 512 * 2;
    f16* w4s  = (f16*)(ws + off); off += (size_t)KTH * 1 * 512 * 2;           //     8,192
    float* preds = (float*)(ws + off); off += (size_t)RTOT * 3 * 4;           // 3,145,728
    float* areas = (float*)(ws + off); off += (size_t)RTOT * 4;               // 1,048,576

    enc_conv<<<(CC * HH * WW + 255) / 256, 256, 0, stream>>>(inp, enc_w, enc_b, feat);

    wswz<<<(KT0 * 16 * 512 + 255) / 256, 256, 0, stream>>>(w0, w0s, INDIM, HID, KT0, 16, 1);
    wswz<<<(KTH * 16 * 512 + 255) / 256, 256, 0, stream>>>(w1, w1s, HID, HID, KTH, 16, 0);
    wswz<<<(KTH * 16 * 512 + 255) / 256, 256, 0, stream>>>(w2, w2s, HID, HID, KTH, 16, 0);
    wswz<<<(KTH * 16 * 512 + 255) / 256, 256, 0, stream>>>(w3, w3s, HID, HID, KTH, 16, 0);
    wswz<<<(KTH * 1 * 512 + 255) / 256, 256, 0, stream>>>(w4, w4s, HID, 3, KTH, 1, 0);

    mlp_kernel<<<RTOT / MTILE, 256, 0, stream>>>(feat, coord, cell,
                                                 w0s, w1s, w2s, w3s, w4s,
                                                 b0, b1, b2, b3, b4, preds, areas);

    combine<<<(QTOT + 255) / 256, 256, 0, stream>>>(preds, areas, out);
}
